// GCNModel_81724637708713
// MI455X (gfx1250) — matrix operand order repaired and checked by reading, not compiled, not run on hardware
//
#include <hip/hip_runtime.h>
#include <hip/hip_bf16.h>

#define N_NODES 65536
#define DIM     64
#define N_EDGES 1048576

typedef __attribute__((ext_vector_type(16))) _Float16 v16h;
typedef __attribute__((ext_vector_type(8)))  float    v8f;

// ---------------- utility kernels ----------------

__global__ void k_zero(float* __restrict__ p, int n) {
    int i = blockIdx.x * blockDim.x + threadIdx.x;
    if (i < n) p[i] = 0.0f;
}

__global__ void k_degrees(const int* __restrict__ src, const int* __restrict__ dst,
                          float* __restrict__ deg_out, float* __restrict__ deg_in) {
    int e = blockIdx.x * blockDim.x + threadIdx.x;
    if (e < N_EDGES) {
        __hip_atomic_fetch_add(&deg_out[src[e]], 1.0f, __ATOMIC_RELAXED, __HIP_MEMORY_SCOPE_AGENT);
        __hip_atomic_fetch_add(&deg_in [dst[e]], 1.0f, __ATOMIC_RELAXED, __HIP_MEMORY_SCOPE_AGENT);
    }
}

// in-place: deg -> deg>0 ? deg^-1/2 : 0
__global__ void k_norms(float* __restrict__ p, int n) {
    int i = blockIdx.x * blockDim.x + threadIdx.x;
    if (i < n) {
        float d = p[i];
        p[i] = (d > 0.0f) ? rsqrtf(d) : 0.0f;
    }
}

// Convert the 3 row-major 64x64 f32 weights into f16, pre-packed in the exact
// per-lane WMMA B-fragment order consumed by k_gemm:
//   pack[(((m*4 + colT)*2 + c)*32 + lane)*16 + j]
//     = W_m[(c*32 + (lane>>4)*8 + (j&7) + (j>>3)*16) * 64 + colT*16 + (lane&15)]
// so each GEMM lane loads its 16 halfs as one contiguous 32-byte run.
__global__ void k_convW(const float* __restrict__ W1, const float* __restrict__ W2,
                        const float* __restrict__ W3, _Float16* __restrict__ Whp) {
    int i = blockIdx.x * blockDim.x + threadIdx.x;
    if (i < 3 * DIM * DIM) {
        int j    = i & 15;
        int lane = (i >> 4) & 31;
        int c    = (i >> 9) & 1;
        int colT = (i >> 10) & 3;
        int m    = i >> 12;
        const float* W = (m == 0) ? W1 : (m == 1 ? W2 : W3);
        int row = c * 32 + (lane >> 4) * 8 + (j & 7) + (j >> 3) * 16;
        int col = colT * 16 + (lane & 15);
        Whp[i] = (_Float16)W[row * DIM + col];
    }
}

// ---------------- WMMA GEMM ----------------
// out[r][c] = sum_k A[r][k] * W[k][c]
//   AFFINE=false: A[r][k] = in[r][k] * ns[r]                       (layer 1)
//   AFFINE=true : A[r][k] = (in[r][k]*alpha[r] + bias[k]) * ns[r]  (layers 2,3;
//                 fuses the previous layer's norm_dst + bias into the A-load)
// One wave per 16x16 output tile; K=64 -> two v_wmma_f32_16x16x32_f16.
template <bool AFFINE>
__global__ __launch_bounds__(256) void k_gemm(
        const float* __restrict__ in, const float* __restrict__ alpha,
        const float* __restrict__ bias, const float* __restrict__ ns,
        const _Float16* __restrict__ Whm, float* __restrict__ out) {
    const int lane = threadIdx.x & 31;
    const int wave = threadIdx.x >> 5;
    const int tile = blockIdx.x * 8 + wave;       // 16384 tiles total
    const int rowT = tile >> 2;                   // 4096 row tiles
    const int colT = tile & 3;                    // 4 col tiles
    const int mrow = lane & 15;
    const int hi   = lane >> 4;                   // lane half (0/1)
    const int row  = rowT * 16 + mrow;
    const int col  = colT * 16 + mrow;            // N index for B/C/D lanes

    const float nsr = ns[row];
    const float al  = AFFINE ? alpha[row] : 1.0f;
    const float* inr = in + row * DIM;

    v8f acc = {};
#pragma unroll
    for (int c = 0; c < 2; ++c) {                 // K chunks: 0..31, 32..63
        // A fragment (ISA 7.12.2, 16-bit A 16x32): lane half selects K base
        // 0/8; a[0..7] <- K abase..abase+7, a[8..15] <- K abase+16..abase+23.
        const int abase = c * 32 + hi * 8;
        const float4 x0 = *(const float4*)(inr + abase);
        const float4 x1 = *(const float4*)(inr + abase + 4);
        const float4 x2 = *(const float4*)(inr + abase + 16);
        const float4 x3 = *(const float4*)(inr + abase + 20);
        float xv[16] = { x0.x, x0.y, x0.z, x0.w,  x1.x, x1.y, x1.z, x1.w,
                         x2.x, x2.y, x2.z, x2.w,  x3.x, x3.y, x3.z, x3.w };
        v16h a;
        if (AFFINE) {
            const float4 b0 = *(const float4*)(bias + abase);
            const float4 b1 = *(const float4*)(bias + abase + 4);
            const float4 b2 = *(const float4*)(bias + abase + 16);
            const float4 b3 = *(const float4*)(bias + abase + 20);
            float bv[16] = { b0.x, b0.y, b0.z, b0.w,  b1.x, b1.y, b1.z, b1.w,
                             b2.x, b2.y, b2.z, b2.w,  b3.x, b3.y, b3.z, b3.w };
#pragma unroll
            for (int j = 0; j < 16; ++j)
                a[j] = (_Float16)((xv[j] * al + bv[j]) * nsr);
        } else {
#pragma unroll
            for (int j = 0; j < 16; ++j)
                a[j] = (_Float16)(xv[j] * nsr);
        }
        // B fragment: pre-packed contiguous 32 bytes per lane.
        const v16h b = *(const v16h*)(Whm + ((colT * 2 + c) * 32 + lane) * 16);
        acc = __builtin_amdgcn_wmma_f32_16x16x32_f16(
                  false, a, false, b, (short)0, acc, false, false);
    }
    // C/D: VGPR v -> M = hi*8 + v, lane -> N (coalesced 64B per half-wave)
#pragma unroll
    for (int v = 0; v < 8; ++v) {
        int M = hi * 8 + v;
        out[(rowT * 16 + M) * DIM + col] = acc[v];
    }
}

// ---------------- edge gather + scatter-add ----------------
// 16 threads per edge; each moves 4 consecutive floats (float4 gather,
// 4x global_atomic_add_f32 scatter).
__global__ void k_scatter(const float* __restrict__ hW, const int* __restrict__ src,
                          const int* __restrict__ dst, float* __restrict__ agg) {
    int tid = blockIdx.x * blockDim.x + threadIdx.x;   // E*16 threads
    int e = tid >> 4;
    int q = tid & 15;
    if (e < N_EDGES) {
        int s = src[e], d = dst[e];
        float4 m = ((const float4*)(hW + (long)s * DIM))[q];
        float* ap = agg + (long)d * DIM + q * 4;
        __hip_atomic_fetch_add(ap + 0, m.x, __ATOMIC_RELAXED, __HIP_MEMORY_SCOPE_AGENT);
        __hip_atomic_fetch_add(ap + 1, m.y, __ATOMIC_RELAXED, __HIP_MEMORY_SCOPE_AGENT);
        __hip_atomic_fetch_add(ap + 2, m.z, __ATOMIC_RELAXED, __HIP_MEMORY_SCOPE_AGENT);
        __hip_atomic_fetch_add(ap + 3, m.w, __ATOMIC_RELAXED, __HIP_MEMORY_SCOPE_AGENT);
    }
}

// out = agg * norm_dst[row] + b[col]  (final layer only)
__global__ void k_finalize(const float* __restrict__ agg, const float* __restrict__ nd,
                           const float* __restrict__ b, float* __restrict__ out) {
    int i = blockIdx.x * blockDim.x + threadIdx.x;     // N*D threads
    int r = i >> 6, c = i & 63;
    out[i] = agg[i] * nd[r] + b[c];
}

// ---------------- launcher ----------------

extern "C" void kernel_launch(void* const* d_in, const int* in_sizes, int n_in,
                              void* d_out, int out_size, void* d_ws, size_t ws_size,
                              hipStream_t stream) {
    const float* x  = (const float*)d_in[0];
    const float* W1 = (const float*)d_in[1];
    const float* b1 = (const float*)d_in[2];
    const float* W2 = (const float*)d_in[3];
    const float* b2 = (const float*)d_in[4];
    const float* W3 = (const float*)d_in[5];
    const float* b3 = (const float*)d_in[6];
    const int*   src = (const int*)d_in[7];
    const int*   dst = (const int*)d_in[8];
    (void)in_sizes; (void)n_in; (void)out_size; (void)ws_size;

    // workspace layout (floats): norms (2N) | hW (N*D) | agg (N*D) | Wh (f16, packed)
    float* ws   = (float*)d_ws;
    float* nsrc = ws;
    float* ndst = ws + N_NODES;
    float* hW   = ws + 2 * N_NODES;
    float* agg  = hW + (long)N_NODES * DIM;
    _Float16* Wh = (_Float16*)(agg + (long)N_NODES * DIM);

    const int T = 256;
    const int gN2   = (2 * N_NODES) / T;           // 512
    const int gE    = N_EDGES / T;                 // 4096
    const int gND   = (N_NODES * DIM) / T;         // 16384
    const int gSc   = (N_EDGES * 16) / T;          // 65536
    const int gGemm = (N_NODES / 16) * 4 / 8;      // 2048 (8 waves/block)

    // degrees -> norms
    k_zero   <<<gN2, T, 0, stream>>>(nsrc, 2 * N_NODES);
    k_degrees<<<gE,  T, 0, stream>>>(src, dst, nsrc, ndst);
    k_norms  <<<gN2, T, 0, stream>>>(nsrc, 2 * N_NODES);
    k_convW  <<<(3 * DIM * DIM) / T, T, 0, stream>>>(W1, W2, W3, Wh);

    // layer 1: A = x * ns
    k_gemm<false><<<gGemm, T, 0, stream>>>(x, nullptr, nullptr, nsrc, Wh, hW);
    k_zero   <<<gND, T, 0, stream>>>(agg, N_NODES * DIM);
    k_scatter<<<gSc, T, 0, stream>>>(hW, src, dst, agg);

    // layer 2: A = (agg * nd + b1) * ns   (prev finalize fused into A-load)
    k_gemm<true><<<gGemm, T, 0, stream>>>(agg, ndst, b1, nsrc, Wh + DIM * DIM, hW);
    k_zero   <<<gND, T, 0, stream>>>(agg, N_NODES * DIM);
    k_scatter<<<gSc, T, 0, stream>>>(hW, src, dst, agg);

    // layer 3
    k_gemm<true><<<gGemm, T, 0, stream>>>(agg, ndst, b2, nsrc, Wh + 2 * DIM * DIM, hW);
    k_zero   <<<gND, T, 0, stream>>>(agg, N_NODES * DIM);
    k_scatter<<<gSc, T, 0, stream>>>(hW, src, dst, agg);

    // final normalize + bias -> output (reshape is a no-op)
    k_finalize<<<gND, T, 0, stream>>>(agg, ndst, b3, (float*)d_out);
}
